// pCL_42442866819451
// MI455X (gfx1250) — matrix operand order repaired and checked by reading, not compiled, not run on hardware
//
#include <hip/hip_runtime.h>
#include <hip/hip_bf16.h>

typedef __attribute__((ext_vector_type(16))) _Float16 v16h;
typedef __attribute__((ext_vector_type(8)))  _Float16 v8h;
typedef __attribute__((ext_vector_type(4)))  _Float16 v4h;
typedef __attribute__((ext_vector_type(8)))  float    v8f;

#define Cn     128          // channels (GEMM M == N)
#define Nn     784          // D*H*W  (= 196 * 4)
#define KP     800          // padded K for GEMM (25 * 32), zeros in [784,800)
#define GSTR   808          // G row stride in f16 (404 dwords -> conflict-free rows)
#define KSTEPS 25           // KP / 32
#define NTILE  36           // upper-triangle tiles of the 8x8 tile grid

// dynamic LDS layout (bytes):
//   s_delta : 128 f32      (  512)
//   s_w     : 800 f32      ( 3200)
//   s_r0    : 256 f32      ( 1024)
//   s_r1    : 256 f32      ( 1024)
//   G       : 128*808 f16  (206848)   -> total 212608 (< 320 KB WGP LDS)
#define SMEM_BYTES ((Cn + KP + 256 + 256) * 4 + Cn * GSTR * 2)

__global__ __launch_bounds__(256)
void pCL_42442866819451_kernel(const float* __restrict__ inp,
                               const float* __restrict__ cent,
                               float* __restrict__ out)
{
    extern __shared__ char smem[];
    float*    s_delta = (float*)smem;
    float*    s_w     = s_delta + Cn;
    float*    s_r0    = s_w + KP;
    float*    s_r1    = s_r0 + 256;
    _Float16* G       = (_Float16*)(s_r1 + 256);   // offset 5760 B, 16B aligned

    const int tid = threadIdx.x;
    const int b   = blockIdx.x;
    const float* __restrict__ inb = inp + (size_t)b * Cn * Nn;

    // ---- centroid delta and bias = ||c1||^2 - ||c0||^2 (deterministic tree reduce)
    float bp = 0.f;
    if (tid < Cn) {
        float c0 = cent[tid * 2 + 0];
        float c1 = cent[tid * 2 + 1];
        s_delta[tid] = c0 - c1;
        bp = c1 * c1 - c0 * c0;
    }
    s_r0[tid] = bp;
    __syncthreads();
    for (int off = 128; off > 0; off >>= 1) {
        if (tid < off) s_r0[tid] += s_r0[tid + off];
        __syncthreads();
    }
    const float bias = s_r0[0];
    __syncthreads();   // protect s_r0 before reuse

    // ---- single HBM pass: stage raw x as f16 into resident LDS G (vectorized)
    for (int i4 = tid; i4 < Cn * (Nn / 4); i4 += 256) {
        int c = i4 / (Nn / 4);
        int n = (i4 - c * (Nn / 4)) * 4;
        const float4 x = *(const float4*)(inb + c * Nn + n);
        v4h h;
        h[0] = (_Float16)x.x; h[1] = (_Float16)x.y;
        h[2] = (_Float16)x.z; h[3] = (_Float16)x.w;
        *(v4h*)(G + c * GSTR + n) = h;
    }
    // zero-fill K tail [784, 808)
    for (int i4 = tid; i4 < Cn * 6; i4 += 256) {
        int c = i4 / 6;
        int q = i4 - c * 6;
        *(v4h*)(G + c * GSTR + Nn + q * 4) = (v4h){(_Float16)0.f, (_Float16)0.f,
                                                   (_Float16)0.f, (_Float16)0.f};
    }
    __syncthreads();

    // ---- masks from LDS: mask0 = sigmoid(2*x.(c0-c1) + bias), mask1 = 1 - mask0
    float p0 = 0.f, p1 = 0.f;
    for (int n = tid; n < KP; n += 256) {
        float wv = 0.f;
        if (n < Nn) {
            float dot = 0.f;
#pragma unroll 8
            for (int c = 0; c < Cn; ++c)
                dot = fmaf((float)G[c * GSTR + n], s_delta[c], dot);
            float t  = 2.f * dot + bias;
            float m0 = __builtin_amdgcn_rcpf(1.f + __expf(-t));   // sigmoid, fast rcp
            float m1 = 1.f - m0;
            p0 += m0;
            p1 += m1;
            wv = __builtin_amdgcn_sqrtf(m0 * m1);
        }
        s_w[n] = wv;
    }
    s_r0[tid] = p0;
    s_r1[tid] = p1;
    __syncthreads();
    for (int off = 128; off > 0; off >>= 1) {
        if (tid < off) {
            s_r0[tid] += s_r0[tid + off];
            s_r1[tid] += s_r1[tid + off];
        }
        __syncthreads();
    }
    // sqrt(s0*s1) with s_k = rsqrt(bsize_k): ((b0+.01)*(b1+.01))^(-1/4)
    const float ssc =
        __builtin_amdgcn_sqrtf(__builtin_amdgcn_rsqf((s_r0[0] + 0.01f) * (s_r1[0] + 0.01f)));

    // ---- in-place scale (vectorized): G[c,n] = x * sqrt(mask0*mask1*s0*s1)
    for (int i4 = tid; i4 < Cn * (KP / 4); i4 += 256) {
        int c = i4 / (KP / 4);
        int n = (i4 - c * (KP / 4)) * 4;
        v4h g = *(v4h*)(G + c * GSTR + n);
        const float4 wv = *(const float4*)(s_w + n);
        g[0] = (_Float16)((float)g[0] * wv.x * ssc);
        g[1] = (_Float16)((float)g[1] * wv.y * ssc);
        g[2] = (_Float16)((float)g[2] * wv.z * ssc);
        g[3] = (_Float16)((float)g[3] * wv.w * ssc);
        *(v4h*)(G + c * GSTR + n) = g;
    }
    __syncthreads();

    // ---- symmetric GEMM: bil = G * G^T, upper-triangle tiles only
    const int lane = tid & 31;
    const int wave = tid >> 5;
    const int l16  = lane & 15;
    const int hh   = lane >> 4;
    float* __restrict__ ob = out + (size_t)b * Cn * Cn;

    for (int t = wave; t < NTILE; t += 8) {
        // decode upper-triangle tile index -> (mi, mj), mi <= mj
        int tt = t, mi = 0;
        while (tt >= 8 - mi) { tt -= 8 - mi; ++mi; }
        const int mj = mi + tt;

        v8f acc = (v8f){0.f,0.f,0.f,0.f,0.f,0.f,0.f,0.f};
        const _Float16* arow = G + (mi * 16 + l16) * GSTR + hh * 8;   // A operand
        const _Float16* brow = G + (mj * 16 + l16) * GSTR + hh * 8;   // B operand

#pragma unroll 5
        for (int ks = 0; ks < KSTEPS; ++ks) {
            const _Float16* ap = arow + ks * 32;
            v8h alo = *(const v8h*)ap;
            v8h ahi = *(const v8h*)(ap + 16);
            const _Float16* bp2 = brow + ks * 32;
            v8h blo = *(const v8h*)bp2;
            v8h bhi = *(const v8h*)(bp2 + 16);
            v16h af, bf;
#pragma unroll
            for (int i = 0; i < 8; ++i) {
                af[i] = alo[i]; af[i + 8] = ahi[i];
                bf[i] = blo[i]; bf[i + 8] = bhi[i];
            }
            acc = __builtin_amdgcn_wmma_f32_16x16x32_f16(
                false, af, false, bf, (short)0, acc, false, false);
        }

        // signed-sqrt epilogue (single v_sqrt_f32); mirror across diagonal if mi != mj
#pragma unroll
        for (int v = 0; v < 8; ++v) {
            int i = mi * 16 + v + hh * 8;
            int j = mj * 16 + l16;
            float a  = acc[v];
            float sg = (a > 0.f) ? 1.f : ((a < 0.f) ? -1.f : 0.f);
            float r  = sg * __builtin_amdgcn_sqrtf(fabsf(a) + 1e-5f);
            ob[i * Cn + j] = r;
            if (mi != mj) ob[j * Cn + i] = r;
        }
    }
}

extern "C" void kernel_launch(void* const* d_in, const int* in_sizes, int n_in,
                              void* d_out, int out_size, void* d_ws, size_t ws_size,
                              hipStream_t stream) {
    (void)in_sizes; (void)n_in; (void)out_size; (void)d_ws; (void)ws_size;
    const float* inp  = (const float*)d_in[0];   // (128,128,1,28,28) f32
    const float* cent = (const float*)d_in[1];   // (1,128,2,1,1,1)   f32
    float* out = (float*)d_out;                  // (128, 128*128)    f32

    static bool attr_set = false;   // host-side only; does not affect stream work
    if (!attr_set) {
        (void)hipFuncSetAttribute((const void*)pCL_42442866819451_kernel,
                                  hipFuncAttributeMaxDynamicSharedMemorySize, SMEM_BYTES);
        attr_set = true;
    }
    pCL_42442866819451_kernel<<<dim3(128), dim3(256), SMEM_BYTES, stream>>>(inp, cent, out);
}
